// Qwen3MoeSparseMoeBlock_6073083756867
// MI455X (gfx1250) — matrix operand order repaired and checked by reading, not compiled, not run on hardware
//
#include <hip/hip_runtime.h>

#define H_DIM 2048
#define I_DIM 768
#define NEXP  16
#define TOPK  8

#define BM 32
#define BN 128
#define BK 32
#define LDK 36   // padded LDS stride (bf16 elems) to dodge bank conflicts

typedef __bf16          v16bf __attribute__((ext_vector_type(16)));
typedef unsigned short  v16us __attribute__((ext_vector_type(16)));
typedef float           v8f   __attribute__((ext_vector_type(8)));

// fp32 -> bf16 round-to-nearest-even
static __device__ __forceinline__ unsigned short f2bf(float f) {
  unsigned u = __float_as_uint(f);
  u += 0x7FFFu + ((u >> 16) & 1u);
  return (unsigned short)(u >> 16);
}

// fast silu(g) * u using v_rcp_f32 (avoids IEEE divide expansion)
static __device__ __forceinline__ float silu_mul(float g, float u) {
  return g * __builtin_amdgcn_rcpf(1.f + __expf(-g)) * u;
}

// CDNA5 async global->LDS copy (4 bytes), tracked by ASYNCcnt.
// lds_off: byte offset within workgroup LDS (low 32 bits of generic pointer).
static __device__ __forceinline__ void async_copy_b32(unsigned lds_off, const void* gaddr) {
  asm volatile("global_load_async_to_lds_b32 %0, %1, off"
               :: "v"(lds_off), "v"(gaddr) : "memory");
}
static __device__ __forceinline__ void wait_asynccnt0() {
  asm volatile("s_wait_asynccnt 0x0" ::: "memory");
}

// A fragment: 16x32 bf16 tile, LDS layout [m][k] stride LDK.
// Lane l<16: row m=l, elems 0..15 -> K 0..15.
// Lane l>=16: row m=l-16, elems 0..15 -> K 16..31.
static __device__ __forceinline__ v16bf load_A(const unsigned short* s, int m0, int lane) {
  const int half = lane >> 4;
  const unsigned short* row = s + (m0 + (lane & 15)) * LDK;
  v16us t;
#pragma unroll
  for (int i = 0; i < 16; ++i) {
    int k = i + 16 * half;
    t[i] = row[k];
  }
  return __builtin_bit_cast(v16bf, t);
}

// B fragment: 32x16 (KxN) bf16 tile, LDS layout [n][k] stride LDK.
// Lane l<16: col n=l, K 0..15; lane l>=16: col n=l-16, K 16..31.
static __device__ __forceinline__ v16bf load_B(const unsigned short* s, int n0, int lane) {
  const unsigned short* col = s + (n0 + (lane & 15)) * LDK + ((lane >> 4) << 4);
  v16us t;
#pragma unroll
  for (int i = 0; i < 16; ++i) t[i] = col[i];
  return __builtin_bit_cast(v16bf, t);
}

static __device__ __forceinline__ v8f bf16_mma(v16bf a, v16bf b, v8f c) {
  return __builtin_amdgcn_wmma_f32_16x16x32_bf16(false, a, false, b, (short)0, c, false, false);
}

// ---------------- Kernel 1: router (logits, softmax, top-8, renorm) -------------
__global__ __launch_bounds__(256) void moe_router(
    const float* __restrict__ X, const float* __restrict__ Wr,
    float* __restrict__ logits, float* __restrict__ dw, int Ntok) {
  const int gid  = blockIdx.x * blockDim.x + threadIdx.x;
  const int tok  = gid >> 5;
  const int lane = threadIdx.x & 31;
  if (tok >= Ntok) return;

  const float* x = X + (size_t)tok * H_DIM;
  float acc[NEXP];
#pragma unroll
  for (int e = 0; e < NEXP; ++e) acc[e] = 0.f;

  for (int h = lane; h < H_DIM; h += 32) {
    float xv = x[h];
    const float* w = Wr + (size_t)h * NEXP;
#pragma unroll
    for (int e = 0; e < NEXP; ++e) acc[e] = fmaf(xv, w[e], acc[e]);
  }
#pragma unroll
  for (int off = 16; off > 0; off >>= 1) {
#pragma unroll
    for (int e = 0; e < NEXP; ++e) acc[e] += __shfl_xor(acc[e], off, 32);
  }
  if (lane < NEXP) logits[(size_t)tok * NEXP + lane] = acc[lane];

  // softmax over 16 (every lane redundantly holds full logits)
  float mx = acc[0];
#pragma unroll
  for (int e = 1; e < NEXP; ++e) mx = fmaxf(mx, acc[e]);
  float p[NEXP], s = 0.f;
#pragma unroll
  for (int e = 0; e < NEXP; ++e) { p[e] = __expf(acc[e] - mx); s += p[e]; }
  float inv = __builtin_amdgcn_rcpf(s);
#pragma unroll
  for (int e = 0; e < NEXP; ++e) p[e] *= inv;

  // top-8 by rank (ties -> lower index wins), renormalize over selected
  float s8 = 0.f;
  int sel[NEXP];
#pragma unroll
  for (int e = 0; e < NEXP; ++e) {
    int r = 0;
#pragma unroll
    for (int j = 0; j < NEXP; ++j)
      r += (p[j] > p[e] || (p[j] == p[e] && j < e)) ? 1 : 0;
    sel[e] = (r < TOPK);
    if (sel[e]) s8 += p[e];
  }
  float inv8 = 1.f / s8;   // exact: feeds the output directly
  if (lane < NEXP) dw[(size_t)tok * NEXP + lane] = sel[lane] ? p[lane] * inv8 : 0.f;
}

// ------------- Kernel 2: act[e][tok][i] = silu(X@Wg) * (X@Wu)  (bf16 out) ------
__global__ __launch_bounds__(256) void moe_gateup(
    const float* __restrict__ X, const float* __restrict__ Wg,
    const float* __restrict__ Wu, unsigned short* __restrict__ act, int Ntok) {
  __shared__ unsigned short Xs[BM * LDK];
  __shared__ unsigned short Gs[BN * LDK];
  __shared__ unsigned short Us[BN * LDK];

  const int t    = threadIdx.x;
  const int lane = t & 31;
  const int wave = t >> 5;
  const int wm   = wave & 1;    // 2 M-subtiles of 16
  const int wn   = wave >> 1;   // 4 N-subtiles of 32

  const int tok0 = blockIdx.x * BM;
  const int i0   = blockIdx.y * BN;
  const int e    = blockIdx.z;

  const float* wg = Wg + (size_t)e * H_DIM * I_DIM;
  const float* wu = Wu + (size_t)e * H_DIM * I_DIM;

  v8f accg[2], accu[2];
#pragma unroll
  for (int s = 0; s < 2; ++s) {
#pragma unroll
    for (int v = 0; v < 8; ++v) { accg[s][v] = 0.f; accu[s][v] = 0.f; }
  }

  for (int k0 = 0; k0 < H_DIM; k0 += BK) {
    // prefetch next K-tile of the weight streams (dominant HBM traffic)
    if (k0 + BK < H_DIM) {
      int kk = t >> 7, ii = t & 127;
      size_t g = (size_t)(k0 + BK + kk) * I_DIM + (i0 + ii);
      __builtin_prefetch(wg + g, 0, 0);
      __builtin_prefetch(wu + g, 0, 0);
    }
    // stage X tile [BM][BK] as bf16, LDS [m][k]
#pragma unroll
    for (int r = 0; r < (BM * BK) / 256; ++r) {
      int idx = t + r * 256;
      int m = idx >> 5, kk = idx & 31;
      Xs[m * LDK + kk] = f2bf(X[(size_t)(tok0 + m) * H_DIM + (k0 + kk)]);
    }
    // stage Wg/Wu tiles [BK][BN] as bf16, transposed into LDS [n][k]
#pragma unroll
    for (int r = 0; r < (BK * BN) / 256; ++r) {
      int idx = t + r * 256;
      int kk = idx >> 7, ii = idx & 127;
      size_t g = (size_t)(k0 + kk) * I_DIM + (i0 + ii);
      Gs[ii * LDK + kk] = f2bf(wg[g]);
      Us[ii * LDK + kk] = f2bf(wu[g]);
    }
    __syncthreads();

    v16bf a = load_A(Xs, wm * 16, lane);
#pragma unroll
    for (int s = 0; s < 2; ++s) {
      v16bf bg = load_B(Gs, wn * 32 + s * 16, lane);
      v16bf bu = load_B(Us, wn * 32 + s * 16, lane);
      accg[s] = bf16_mma(a, bg, accg[s]);
      accu[s] = bf16_mma(a, bu, accu[s]);
    }
    __syncthreads();
  }

  // silu(gate) * up -> bf16 act, scattered per C-fragment layout
  const int half = lane >> 4, nc = lane & 15;
#pragma unroll
  for (int s = 0; s < 2; ++s) {
#pragma unroll
    for (int v = 0; v < 8; ++v) {
      float a = silu_mul(accg[s][v], accu[s][v]);
      int m = wm * 16 + v + 8 * half;
      int i = i0 + wn * 32 + s * 16 + nc;
      act[((size_t)e * Ntok + (tok0 + m)) * I_DIM + i] = f2bf(a);
    }
  }
}

// ------- Kernel 3: final[tok][h] = sum_e dw[tok,e] * (act[e,tok,:] @ Wd[e]) -----
__global__ __launch_bounds__(256) void moe_down(
    const unsigned short* __restrict__ act, const float* __restrict__ Wd,
    const float* __restrict__ dw, float* __restrict__ out, int Ntok) {
  __shared__ unsigned short As[BM * LDK];
  __shared__ unsigned short Ds[BN * LDK];
  __shared__ float dws[BM * NEXP];

  const int t    = threadIdx.x;
  const int lane = t & 31;
  const int wave = t >> 5;
  const int wm   = wave & 1;
  const int wn   = wave >> 1;
  const int half = lane >> 4;

  const int tok0 = blockIdx.x * BM;
  const int h0   = blockIdx.y * BN;

  // stage combine weights [BM][NEXP] (contiguous rows)
#pragma unroll
  for (int r = 0; r < (BM * NEXP) / 256; ++r) {
    int idx = t + r * 256;
    dws[idx] = dw[(size_t)tok0 * NEXP + idx];
  }

  v8f fin[2];
#pragma unroll
  for (int s = 0; s < 2; ++s)
#pragma unroll
    for (int v = 0; v < 8; ++v) fin[s][v] = 0.f;

  for (int e = 0; e < NEXP; ++e) {
    v8f ac[2];
#pragma unroll
    for (int s = 0; s < 2; ++s)
#pragma unroll
      for (int v = 0; v < 8; ++v) ac[s][v] = 0.f;

    const float* wd = Wd + (size_t)e * I_DIM * H_DIM;
    const unsigned short* ae = act + (size_t)e * Ntok * I_DIM;

    for (int k0 = 0; k0 < I_DIM; k0 += BK) {
      __syncthreads();   // LDS reads of previous tile complete
      // prefetch next Wd K-tile
      if (k0 + BK < I_DIM) {
        int kk = t >> 7, nn = t & 127;
        __builtin_prefetch(wd + (size_t)(k0 + BK + kk) * H_DIM + (h0 + nn), 0, 0);
      }
      // stage act tile [BM][BK] bf16 via async global->LDS copies (pure byte copy,
      // no VGPR round-trip; tracked by ASYNCcnt). 512 dwords, 2 per thread.
#pragma unroll
      for (int r = 0; r < (BM * BK * 2) / (256 * 4); ++r) {  // dwords / threads
        int d = t + r * 256;
        int m = d >> 4, kp = (d & 15) << 1;                  // pair of bf16
        unsigned lds_off = (unsigned)(unsigned long long)(const void*)&As[m * LDK + kp];
        const void* g = (const void*)&ae[(size_t)(tok0 + m) * I_DIM + (k0 + kp)];
        async_copy_b32(lds_off, g);
      }
      // stage Wd tile [BK][BN] fp32->bf16 into LDS [n][k]
#pragma unroll
      for (int r = 0; r < (BK * BN) / 256; ++r) {
        int idx = t + r * 256;
        int kk = idx >> 7, nn = idx & 127;
        Ds[nn * LDK + kk] = f2bf(wd[(size_t)(k0 + kk) * H_DIM + (h0 + nn)]);
      }
      wait_asynccnt0();  // this wave's async LDS writes landed
      __syncthreads();   // all waves' tiles visible

      v16bf a = load_A(As, wm * 16, lane);
#pragma unroll
      for (int s = 0; s < 2; ++s) {
        v16bf b = load_B(Ds, wn * 32 + s * 16, lane);
        ac[s] = bf16_mma(a, b, ac[s]);
      }
    }
    // scale by per-token combine weight and accumulate
#pragma unroll
    for (int v = 0; v < 8; ++v) {
      int m = wm * 16 + v + 8 * half;
      float w = dws[m * NEXP + e];
      fin[0][v] = fmaf(w, ac[0][v], fin[0][v]);
      fin[1][v] = fmaf(w, ac[1][v], fin[1][v]);
    }
  }

  const int nc = lane & 15;
#pragma unroll
  for (int s = 0; s < 2; ++s) {
#pragma unroll
    for (int v = 0; v < 8; ++v) {
      int m = wm * 16 + v + 8 * half;
      int h = h0 + wn * 32 + s * 16 + nc;
      out[(size_t)(tok0 + m) * H_DIM + h] = fin[s][v];
    }
  }
}

extern "C" void kernel_launch(void* const* d_in, const int* in_sizes, int n_in,
                              void* d_out, int out_size, void* d_ws, size_t ws_size,
                              hipStream_t stream) {
  (void)n_in; (void)out_size; (void)ws_size;
  const float* X  = (const float*)d_in[0];
  const float* Wr = (const float*)d_in[1];
  const float* Wg = (const float*)d_in[2];
  const float* Wu = (const float*)d_in[3];
  const float* Wd = (const float*)d_in[4];

  const int Ntok = in_sizes[0] / H_DIM;   // B*S = 4096

  float* out_final  = (float*)d_out;                       // [Ntok, H]
  float* out_logits = out_final + (size_t)Ntok * H_DIM;    // [Ntok, E]

  float* dwbuf = (float*)d_ws;                                        // [Ntok, E] fp32
  size_t dw_bytes = (((size_t)Ntok * NEXP * sizeof(float)) + 255) & ~(size_t)255;
  unsigned short* actbuf = (unsigned short*)((char*)d_ws + dw_bytes); // [E, Ntok, I] bf16

  // K1: router
  {
    int threads = Ntok * 32;
    moe_router<<<(threads + 255) / 256, 256, 0, stream>>>(X, Wr, out_logits, dwbuf, Ntok);
  }
  // K2: gate/up + silu*up  (expert on grid-z so each expert's weights stream once through L2)
  {
    dim3 grid(Ntok / BM, I_DIM / BN, NEXP);
    moe_gateup<<<grid, 256, 0, stream>>>(X, Wg, Wu, actbuf, Ntok);
  }
  // K3: down projection + weighted combine (register accumulation, no atomics)
  {
    dim3 grid(Ntok / BM, H_DIM / BN);
    moe_down<<<grid, 256, 0, stream>>>(actbuf, Wd, dwbuf, out_final, Ntok);
  }
}
